// DynamicConvolution_76184129896915
// MI455X (gfx1250) — matrix operand order repaired and checked by reading, not compiled, not run on hardware
//
#include <hip/hip_runtime.h>

#define BB 8
#define CC 64
#define HH 128
#define WW 128
#define TN 32          // pixels per workgroup tile
#define FPAD 33        // padded row stride for f in LDS

typedef __attribute__((ext_vector_type(16))) __bf16 v16bf;
typedef __attribute__((ext_vector_type(8)))  __bf16 v8bf;
typedef __attribute__((ext_vector_type(8)))  float  v8f;

__device__ __forceinline__ unsigned short f2bf(float f) {
  unsigned int u = __float_as_uint(f);
  unsigned int r = (u + 0x7FFFu + ((u >> 16) & 1u)) >> 16;  // round-to-nearest-even
  return (unsigned short)r;
}

// One-time conversion of w_gen (576x64 f32) -> bf16 in workspace.
__global__ void wgen_to_bf16(const float* __restrict__ w,
                             unsigned short* __restrict__ wbf, int n) {
  int i = blockIdx.x * 256 + threadIdx.x;
  if (i < n) wbf[i] = f2bf(w[i]);
}

__global__ __launch_bounds__(256)
void dynconv_kernel(const float* __restrict__ x,
                    const float* __restrict__ bgen,
                    const unsigned short* __restrict__ wbf,
                    float* __restrict__ out) {
  extern __shared__ char smem[];
  float* xs  = (float*)smem;                              // [CC][3][TN+2] f32
  float* bgs = xs + CC * 3 * (TN + 2);                    // [576]
  unsigned short* Bsm = (unsigned short*)(bgs + 9 * CC);  // [TN][CC] bf16
  float* fs  = (float*)(Bsm + TN * CC);                   // [576][FPAD]

  const int tid = threadIdx.x;
  const int w0  = blockIdx.x * TN;
  const int h   = blockIdx.y;
  const int b   = blockIdx.z;
  const float* xb = x + (size_t)b * CC * HH * WW;

  // ---- Stage 1: stage x tile (rows h-1,h,h+1 with w halo, zero-padded) + bias.
  //      Center-row interior values are also written to the bf16 B matrix directly.
  const int XSN = CC * 3 * (TN + 2);
  for (int i = tid; i < XSN; i += 256) {
    int c   = i / (3 * (TN + 2));
    int rem = i - c * 3 * (TN + 2);
    int r   = rem / (TN + 2);
    int col = rem - r * (TN + 2);
    int hh  = h - 1 + r;
    int ww  = w0 - 1 + col;
    float v = 0.0f;
    if ((unsigned)hh < HH && (unsigned)ww < WW)
      v = xb[(size_t)c * HH * WW + (size_t)hh * WW + ww];
    xs[i] = v;
    if (r == 1 && (unsigned)(col - 1) < (unsigned)TN)
      Bsm[(col - 1) * CC + c] = f2bf(v);   // B matrix [pixel][channel] bf16
  }
  for (int i = tid; i < 9 * CC; i += 256) bgs[i] = bgen[i];
  __syncthreads();

  // ---- Stage 2: f(576 x TN) = Wgen(576x64) * X(64xTN), bf16 WMMA, f32 acc ----
  const int wave = tid >> 5;
  const int lane = tid & 31;
  const int lrow = lane & 15;   // M (A) / N (B) index within 16x16 tile
  const int lhi  = lane >> 4;   // lane-half selects K interleave
  // Tile t = wave + 8*j  (j = 0..8):  Nt = wave&1 (invariant), Mt = (wave>>1) + 4*j.
  const int Nt  = wave & 1;
  const int Mt0 = wave >> 1;
  const int p   = Nt * 16 + lrow;          // pixel (B N index)

  // B fragments are loop-invariant per wave: hoist both K-steps.
  // Lanes 0-15 hold K {0..7,16..23}, lanes 16-31 hold K {8..15,24..31} (matches A).
  const __bf16* bbase = (const __bf16*)Bsm + p * CC + lhi * 8;
  const v16bf bm0 = __builtin_shufflevector(*(const v8bf*)(bbase), *(const v8bf*)(bbase + 16),
      0,1,2,3,4,5,6,7,8,9,10,11,12,13,14,15);
  const v16bf bm1 = __builtin_shufflevector(*(const v8bf*)(bbase + 32), *(const v8bf*)(bbase + 48),
      0,1,2,3,4,5,6,7,8,9,10,11,12,13,14,15);

  // A fragments: lanes 0-15 hold K {0..7,16..23}, lanes 16-31 hold K {8..15,24..31}.
  // Tiles are software-pipelined one iteration ahead to hide the w_gen load latency.
  const __bf16* wrow = (const __bf16*)wbf + (size_t)(Mt0 * 16 + lrow) * CC + lhi * 8;
  const int WSTRIDE = 64 * CC;             // elements between tile j and j+1 (4 M-tiles)

  v8bf c0 = *(const v8bf*)(wrow);
  v8bf c1 = *(const v8bf*)(wrow + 16);
  v8bf c2 = *(const v8bf*)(wrow + 32);
  v8bf c3 = *(const v8bf*)(wrow + 48);

#pragma unroll
  for (int j = 0; j < 9; ++j) {            // uniform trip count: no exec masking
    v8bf n0, n1, n2, n3;
    if (j < 8) {                           // resolved at unroll time
      const __bf16* nrow = wrow + (size_t)(j + 1) * WSTRIDE;
      n0 = *(const v8bf*)(nrow);
      n1 = *(const v8bf*)(nrow + 16);
      n2 = *(const v8bf*)(nrow + 32);
      n3 = *(const v8bf*)(nrow + 48);
    }
    v16bf am0 = __builtin_shufflevector(c0, c1,
        0,1,2,3,4,5,6,7,8,9,10,11,12,13,14,15);
    v16bf am1 = __builtin_shufflevector(c2, c3,
        0,1,2,3,4,5,6,7,8,9,10,11,12,13,14,15);
    v8f acc = {};
    acc = __builtin_amdgcn_wmma_f32_16x16x32_bf16(
        false, am0, false, bm0, (short)0, acc, false, false);
    acc = __builtin_amdgcn_wmma_f32_16x16x32_bf16(
        false, am1, false, bm1, (short)0, acc, false, false);
    // D layout: VGPR r -> M = r (lanes 0-15) / r+8 (lanes 16-31), N = lane%16
    const int Mt = Mt0 + 4 * j;
#pragma unroll
    for (int r = 0; r < 8; ++r)
      fs[(Mt * 16 + lhi * 8 + r) * FPAD + Nt * 16 + lrow] = acc[r];
    c0 = n0; c1 = n1; c2 = n2; c3 = n3;
  }
  __syncthreads();

  // ---- Stage 3: out[c,p] = sum_k (f[c*9+k,p] + b[c*9+k]) * x[c, h+kr-1, w0+p+kc-1] ----
  const int pp = tid & (TN - 1);
  const int cg = tid >> 5;
  float* orow = out + ((size_t)b * CC) * HH * WW + (size_t)h * WW + w0;
#pragma unroll
  for (int j = 0; j < 8; ++j) {
    int c = cg * 8 + j;
    const float* fcol = fs + (c * 9) * FPAD + pp;
    const float* bg   = bgs + c * 9;
    const float* xc   = xs + c * 3 * (TN + 2);
    float sum = 0.0f;
#pragma unroll
    for (int kr = 0; kr < 3; ++kr)
#pragma unroll
      for (int kc = 0; kc < 3; ++kc) {
        int k = kr * 3 + kc;
        sum += (fcol[k * FPAD] + bg[k]) * xc[kr * (TN + 2) + pp + kc];
      }
    orow[(size_t)c * HH * WW + pp] = sum;
  }
}

extern "C" void kernel_launch(void* const* d_in, const int* in_sizes, int n_in,
                              void* d_out, int out_size, void* d_ws, size_t ws_size,
                              hipStream_t stream) {
  const float* x     = (const float*)d_in[0];
  const float* w_gen = (const float*)d_in[1];
  const float* b_gen = (const float*)d_in[2];
  float* out = (float*)d_out;

  unsigned short* wbf = (unsigned short*)d_ws;   // 576*64 bf16 = 73728 B
  const int n = 9 * CC * CC;                     // 36864
  wgen_to_bf16<<<(n + 255) / 256, 256, 0, stream>>>(w_gen, wbf, n);

  dim3 grid(WW / TN, HH, BB);                    // 4 x 128 x 8
  size_t shmem = (size_t)(CC * 3 * (TN + 2) + 9 * CC) * sizeof(float)   // xs + bias
               + (size_t)(TN * CC) * sizeof(unsigned short)             // B bf16
               + (size_t)(9 * CC * FPAD) * sizeof(float);               // f
  dynconv_kernel<<<grid, 256, shmem, stream>>>(x, b_gen, wbf, out);
}
